// PoseBiLSTM_11261404250889
// MI455X (gfx1250) — matrix operand order repaired and checked by reading, not compiled, not run on hardware
//
#include <hip/hip_runtime.h>
#include <hip/hip_bf16.h>

// ---------------------------------------------------------------------------
// PoseBiLSTM on MI455X (gfx1250): bf16 WMMA for conv-as-GEMM and the fused
// (input-projection + recurrence) BiLSTM scan; VALU kernels for the tail.
// Conv GEMM is double-buffered in LDS (HBM-bound: ~200MB streamed once) and
// uses global_prefetch_b8 two K-chunks ahead; the LSTM scan prefetches the
// next timestep's activations under the current step's WMMAs.
// ---------------------------------------------------------------------------

#define B_  256
#define T_  512
#define F_  378
#define C_  128   // conv out channels == lstm input
#define H_  64
#define G_  256   // 4*H
#define KC_ 1152  // conv GEMM K padded: 3 taps * 384 (378 padded to 384)

typedef __attribute__((ext_vector_type(16))) __bf16 bf16x16;
typedef __attribute__((ext_vector_type(8)))  float  floatx8;

union FragU { bf16x16 v; unsigned u[8]; unsigned short h[16]; };

__device__ __forceinline__ unsigned short f2bf(float f) {
  unsigned u = __float_as_uint(f);
  u += 0x7FFFu + ((u >> 16) & 1u);        // round-to-nearest-even
  return (unsigned short)(u >> 16);
}

// A fragment (16x32 bf16, M x K) from row-major LDS tile (element stride SA).
// Lanes 0-15: row M=lane, K=kb..kb+15;
// lanes 16-31: row M=lane-16, K=kb+16..kb+31.
__device__ __forceinline__ bf16x16 load_frag_a(const unsigned short* s,
                                               int mrow, int kb, int SA) {
  const int lane = threadIdx.x & 31;
  const int r = lane & 15, half = lane >> 4;
  FragU f;
  const unsigned* p = (const unsigned*)(s + (mrow + r) * SA + kb + 16 * half);
#pragma unroll
  for (int i = 0; i < 8; ++i) f.u[i] = p[i];
  return f.v;
}

// B fragment (32x16 bf16, K x N) from n-major LDS tile s[n][k] (stride SB).
// Lanes 0-15: col N=lane, K=kb..kb+15; lanes 16-31: col N=lane-16, K=kb+16..kb+31.
__device__ __forceinline__ bf16x16 load_frag_b(const unsigned short* s,
                                               int ncol, int kb, int SB) {
  const int lane = threadIdx.x & 31;
  const int r = lane & 15, half = lane >> 4;
  FragU f;
  const unsigned* p = (const unsigned*)(s + (ncol + r) * SB + kb + 16 * half);
#pragma unroll
  for (int i = 0; i < 8; ++i) f.u[i] = p[i];
  return f.v;
}

__device__ __forceinline__ floatx8 wmma_bf16(bf16x16 a, bf16x16 b, floatx8 c) {
  return __builtin_amdgcn_wmma_f32_16x16x32_bf16(false, a, false, b,
                                                 (short)0, c, false, false);
}

__device__ __forceinline__ float sigmoidf_(float x) {
  return 1.0f / (1.0f + __expf(-x));
}

// ---------------------------------------------------------------------------
// Kernel 1: Conv1d(F->C, k=3, pad=1) + bias + ReLU as implicit GEMM.
//   A[m, kk] = x[b, t+tap-1, f]  (m=b*T+t, kk=tap*384+f, zero-padded)
//   B[kk, c] = conv_w[c, f, tap]
// Output y in bf16 [M, C]. 1024 blocks x 256 threads (8 waves); each wave owns
// one 16-row M strip x 8 N tiles. 36 K chunks of 32, double-buffered in LDS.
// ---------------------------------------------------------------------------
__global__ __launch_bounds__(256) void conv_wmma_kernel(
    const float* __restrict__ x, const float* __restrict__ conv_w,
    const float* __restrict__ conv_b, unsigned short* __restrict__ yb) {
  __shared__ unsigned short sA[2][128 * 36];
  __shared__ unsigned short sB[2][128 * 36];
  const int blockM = blockIdx.x * 128;
  const int tid = threadIdx.x;
  const int wave = tid >> 5;

  floatx8 acc[8];
#pragma unroll
  for (int j = 0; j < 8; ++j)
#pragma unroll
    for (int e = 0; e < 8; ++e) acc[j][e] = 0.0f;

  auto stage = [&](int buf, int kb) {
    // Stage A: 128 rows x 32 k, gather conv taps, f32 -> bf16.
    for (int i = tid; i < 128 * 32; i += 256) {
      const int row = i >> 5, kc = i & 31;
      const int m = blockM + row;
      const int b = m >> 9, t = m & 511;           // T_ == 512
      const int kk = kb + kc;
      const int tap = kk / 384, f = kk - tap * 384;
      const int ts = t + tap - 1;
      float v = 0.0f;
      if (f < F_ && (unsigned)ts < (unsigned)T_) v = x[(b * T_ + ts) * F_ + f];
      sA[buf][row * 36 + kc] = f2bf(v);
    }
    // Stage B (n-major): sB[n][kc] = conv_w[n, f, tap].
    for (int i = tid; i < 128 * 32; i += 256) {
      const int n = i >> 5, kc = i & 31;
      const int kk = kb + kc;
      const int tap = kk / 384, f = kk - tap * 384;
      float v = 0.0f;
      if (f < F_) v = conv_w[(n * F_ + f) * 3 + tap];
      sB[buf][n * 36 + kc] = f2bf(v);
    }
  };

  const int nch = KC_ / 32;
  stage(0, 0);
  for (int c = 0; c < nch; ++c) {
    __syncthreads();
    if (c + 1 < nch) stage((c + 1) & 1, (c + 1) * 32);

    // Prefetch the A-source cachelines two chunks ahead (global_prefetch_b8).
    if (c + 2 < nch) {
      const int kb2 = (c + 2) * 32;
      const int row = tid >> 1, kc = (tid & 1) * 16;   // 2 lines per row
      const int m = blockM + row;
      const int b = m >> 9, t = m & 511;
      const int kk = kb2 + kc;
      const int tap = kk / 384, f = kk - tap * 384;
      const int ts = t + tap - 1;
      if (f < F_ && (unsigned)ts < (unsigned)T_)
        __builtin_prefetch(&x[(b * T_ + ts) * F_ + f], 0, 1);
    }

    const unsigned short* Abuf = sA[c & 1];
    const unsigned short* Bbuf = sB[c & 1];
    const bf16x16 a = load_frag_a(Abuf, wave * 16, 0, 36);
#pragma unroll
    for (int j = 0; j < 8; ++j) {
      const bf16x16 b = load_frag_b(Bbuf, j * 16, 0, 36);
      acc[j] = wmma_bf16(a, b, acc[j]);
    }
  }

  // Epilogue: bias + ReLU, store bf16. C layout: M = 8*half + vgpr, N = lane&15.
  const int lane = tid & 31, half = lane >> 4, col = lane & 15;
#pragma unroll
  for (int j = 0; j < 8; ++j) {
#pragma unroll
    for (int rv = 0; rv < 8; ++rv) {
      const int m = blockM + wave * 16 + half * 8 + rv;
      const int n = j * 16 + col;
      float v = acc[j][rv] + conv_b[n];
      v = v > 0.0f ? v : 0.0f;
      yb[m * C_ + n] = f2bf(v);
    }
  }
}

// ---------------------------------------------------------------------------
// Kernel 2: fused BiLSTM scan. Grid = 32 blocks: dir = blk>>4, batch-slice of
// 16 rows = (blk&15)*16. 128 threads = 4 waves; wave w owns gate type w
// (i,f,g,o), i.e. gate columns [64w, 64w+64). Per timestep:
//   gates = y_t[16x128] @ w_ih^T + h[16x64] @ w_hh^T + bias  (24 WMMAs/wave)
// then activations + elementwise c/h update through LDS. Next timestep's
// activation tile is prefetched under the WMMAs (the scan serializes, so this
// latency is otherwise exposed).
// ---------------------------------------------------------------------------
__global__ __launch_bounds__(128) void lstm_wmma_kernel(
    const unsigned short* __restrict__ yb,
    const float* __restrict__ w_ih_f, const float* __restrict__ w_hh_f,
    const float* __restrict__ b_f,
    const float* __restrict__ w_ih_b, const float* __restrict__ w_hh_b,
    const float* __restrict__ b_b, float* __restrict__ hout) {
  __shared__ unsigned short s_wih[G_ * C_];  // 64KB, n-major [gate][k]
  __shared__ unsigned short s_whh[G_ * H_];  // 32KB, n-major [gate][k]
  __shared__ unsigned short s_y[16 * C_];    //  4KB, row-major [m][k]
  __shared__ unsigned short s_h[16 * H_];    //  2KB, row-major [m][k]
  __shared__ float s_c[16 * H_];             //  4KB
  __shared__ float s_g[16 * G_];             // 16KB activated gates
  __shared__ float s_bias[G_];

  const int dir = blockIdx.x >> 4;
  const int b0 = (blockIdx.x & 15) * 16;
  const int tid = threadIdx.x;
  const int wave = tid >> 5;
  const float* wih = dir ? w_ih_b : w_ih_f;
  const float* whh = dir ? w_hh_b : w_hh_f;
  const float* bia = dir ? b_b : b_f;

  for (int i = tid; i < G_ * C_; i += 128) s_wih[i] = f2bf(wih[i]);
  for (int i = tid; i < G_ * H_; i += 128) s_whh[i] = f2bf(whh[i]);
  for (int i = tid; i < G_; i += 128) s_bias[i] = bia[i];
  for (int i = tid; i < 16 * H_; i += 128) { s_h[i] = 0; s_c[i] = 0.0f; }
  __syncthreads();

  const int lane = tid & 31, half = lane >> 4, col = lane & 15;

  for (int it = 0; it < T_; ++it) {
    const int t = dir ? (T_ - 1 - it) : it;

    // Stage this timestep's conv activations for the batch slice.
    for (int i = tid; i < 16 * C_; i += 128) {
      const int m = i >> 7, k = i & 127;
      s_y[i] = yb[((b0 + m) * T_ + t) * C_ + k];
    }
    __syncthreads();

    // Prefetch next timestep's activation rows (hidden under the WMMAs).
    if (it + 1 < T_) {
      const int tn = dir ? (t - 1) : (t + 1);
      const int m = tid >> 3, seg = (tid & 7) * 16;   // 8 x 32B per row
      __builtin_prefetch(&yb[((b0 + m) * T_ + tn) * C_ + seg], 0, 1);
    }

    floatx8 acc[4];
#pragma unroll
    for (int j = 0; j < 4; ++j)
#pragma unroll
      for (int e = 0; e < 8; ++e) acc[j][e] = 0.0f;

    // Input projection: K = 128.
#pragma unroll
    for (int kb = 0; kb < C_; kb += 32) {
      const bf16x16 a = load_frag_a(s_y, 0, kb, C_);
#pragma unroll
      for (int j = 0; j < 4; ++j) {
        const bf16x16 b = load_frag_b(s_wih, wave * 64 + j * 16, kb, C_);
        acc[j] = wmma_bf16(a, b, acc[j]);
      }
    }
    // Recurrent term: K = 64.
#pragma unroll
    for (int kb = 0; kb < H_; kb += 32) {
      const bf16x16 a = load_frag_a(s_h, 0, kb, H_);
#pragma unroll
      for (int j = 0; j < 4; ++j) {
        const bf16x16 b = load_frag_b(s_whh, wave * 64 + j * 16, kb, H_);
        acc[j] = wmma_bf16(a, b, acc[j]);
      }
    }

    // Activate: wave 2 holds gate g (tanh); waves 0,1,3 hold i,f,o (sigmoid).
#pragma unroll
    for (int j = 0; j < 4; ++j) {
#pragma unroll
      for (int rv = 0; rv < 8; ++rv) {
        const int mrow = half * 8 + rv;
        const int n = wave * 64 + j * 16 + col;
        const float raw = acc[j][rv] + s_bias[n];
        const float av = (wave == 2) ? tanhf(raw) : sigmoidf_(raw);
        s_g[mrow * G_ + n] = av;
      }
    }
    __syncthreads();

    // c/h update: 1024 elements, 8 per thread.
#pragma unroll
    for (int e = 0; e < 8; ++e) {
      const int idx = tid * 8 + e;
      const int m = idx >> 6, hh = idx & 63;
      const float iv = s_g[m * G_ + hh];
      const float fv = s_g[m * G_ + 64 + hh];
      const float gv = s_g[m * G_ + 128 + hh];
      const float ov = s_g[m * G_ + 192 + hh];
      const float c = fv * s_c[m * H_ + hh] + iv * gv;
      const float h = ov * tanhf(c);
      s_c[m * H_ + hh] = c;
      s_h[m * H_ + hh] = f2bf(h);
      hout[((size_t)(b0 + m) * T_ + t) * (2 * H_) + dir * H_ + hh] = h;
    }
    __syncthreads();
  }
}

// ---------------------------------------------------------------------------
// Kernel 3: attention logits[b,t] = <hout[b,t,:], attn_w> + attn_b.
// ---------------------------------------------------------------------------
__global__ __launch_bounds__(256) void attn_logits_kernel(
    const float* __restrict__ hout, const float* __restrict__ attn_w,
    const float* __restrict__ attn_b, float* __restrict__ logits) {
  const int bt = blockIdx.x * 256 + threadIdx.x;
  const float* row = hout + (size_t)bt * (2 * H_);
  float s = attn_b[0];
#pragma unroll 4
  for (int d = 0; d < 2 * H_; ++d) s += row[d] * attn_w[d];
  logits[bt] = s;
}

// ---------------------------------------------------------------------------
// Kernel 4: per-batch softmax over T, attention pooling, LayerNorm, FC(2).
// One block (256 threads) per batch row.
// ---------------------------------------------------------------------------
__global__ __launch_bounds__(256) void pool_ln_fc_kernel(
    const float* __restrict__ hout, const float* __restrict__ logits,
    const float* __restrict__ ln_g, const float* __restrict__ ln_b,
    const float* __restrict__ fc_w, const float* __restrict__ fc_b,
    float* __restrict__ out) {
  __shared__ float red[256];
  __shared__ float s_w[T_];
  __shared__ float s_pool[2 * H_];
  __shared__ float s_mu, s_var;
  const int b = blockIdx.x, tid = threadIdx.x;
  const float* lg = logits + (size_t)b * T_;

  // softmax max
  const float l0 = lg[tid], l1 = lg[tid + 256];
  red[tid] = fmaxf(l0, l1);
  __syncthreads();
  for (int s = 128; s > 0; s >>= 1) {
    if (tid < s) red[tid] = fmaxf(red[tid], red[tid + s]);
    __syncthreads();
  }
  const float mx = red[0];
  __syncthreads();
  // softmax sum
  const float e0 = __expf(l0 - mx), e1 = __expf(l1 - mx);
  red[tid] = e0 + e1;
  __syncthreads();
  for (int s = 128; s > 0; s >>= 1) {
    if (tid < s) red[tid] += red[tid + s];
    __syncthreads();
  }
  const float inv = 1.0f / red[0];
  __syncthreads();
  s_w[tid] = e0 * inv;
  s_w[tid + 256] = e1 * inv;
  __syncthreads();

  // pooled[d] = sum_t w[t] * hout[b,t,d]
  if (tid < 2 * H_) {
    float a = 0.0f;
    for (int t = 0; t < T_; ++t)
      a += s_w[t] * hout[((size_t)b * T_ + t) * (2 * H_) + tid];
    s_pool[tid] = a;
  }
  __syncthreads();

  // LayerNorm stats over 128 dims.
  const float pv = (tid < 2 * H_) ? s_pool[tid] : 0.0f;
  red[tid] = pv;
  __syncthreads();
  for (int s = 128; s > 0; s >>= 1) {
    if (tid < s) red[tid] += red[tid + s];
    __syncthreads();
  }
  if (tid == 0) s_mu = red[0] / (float)(2 * H_);
  __syncthreads();
  const float d0 = pv - s_mu;
  red[tid] = (tid < 2 * H_) ? d0 * d0 : 0.0f;
  __syncthreads();
  for (int s = 128; s > 0; s >>= 1) {
    if (tid < s) red[tid] += red[tid + s];
    __syncthreads();
  }
  if (tid == 0) s_var = red[0] / (float)(2 * H_);
  __syncthreads();

  const float nv = (tid < 2 * H_)
      ? d0 * rsqrtf(s_var + 1e-5f) * ln_g[tid] + ln_b[tid] : 0.0f;

  // FC: two output logits.
  red[tid] = (tid < 2 * H_) ? nv * fc_w[tid] : 0.0f;
  __syncthreads();
  for (int s = 128; s > 0; s >>= 1) {
    if (tid < s) red[tid] += red[tid + s];
    __syncthreads();
  }
  if (tid == 0) out[b * 2 + 0] = red[0] + fc_b[0];
  __syncthreads();
  red[tid] = (tid < 2 * H_) ? nv * fc_w[2 * H_ + tid] : 0.0f;
  __syncthreads();
  for (int s = 128; s > 0; s >>= 1) {
    if (tid < s) red[tid] += red[tid + s];
    __syncthreads();
  }
  if (tid == 0) out[b * 2 + 1] = red[0] + fc_b[1];
}

// ---------------------------------------------------------------------------
// Host-side launch. Workspace: yb (bf16, 33.5MB) | hout (f32, 67MB) |
// logits (f32, 0.5MB) ~ 101MB total.
// ---------------------------------------------------------------------------
extern "C" void kernel_launch(void* const* d_in, const int* in_sizes, int n_in,
                              void* d_out, int out_size, void* d_ws,
                              size_t ws_size, hipStream_t stream) {
  const float* x      = (const float*)d_in[0];
  const float* conv_w = (const float*)d_in[1];
  const float* conv_b = (const float*)d_in[2];
  const float* w_ih_f = (const float*)d_in[3];
  const float* w_hh_f = (const float*)d_in[4];
  const float* b_f    = (const float*)d_in[5];
  const float* w_ih_b = (const float*)d_in[6];
  const float* w_hh_b = (const float*)d_in[7];
  const float* b_b    = (const float*)d_in[8];
  const float* attn_w = (const float*)d_in[9];
  const float* attn_b = (const float*)d_in[10];
  const float* ln_g   = (const float*)d_in[11];
  const float* ln_b   = (const float*)d_in[12];
  const float* fc_w   = (const float*)d_in[13];
  const float* fc_b   = (const float*)d_in[14];

  char* ws = (char*)d_ws;
  const size_t yb_bytes   = (size_t)B_ * T_ * C_ * sizeof(unsigned short);
  const size_t hout_bytes = (size_t)B_ * T_ * 2 * H_ * sizeof(float);
  unsigned short* yb = (unsigned short*)ws;
  float* hout   = (float*)(ws + yb_bytes);
  float* logits = (float*)(ws + yb_bytes + hout_bytes);
  float* out = (float*)d_out;

  conv_wmma_kernel<<<(B_ * T_) / 128, 256, 0, stream>>>(x, conv_w, conv_b, yb);
  lstm_wmma_kernel<<<32, 128, 0, stream>>>(yb, w_ih_f, w_hh_f, b_f,
                                           w_ih_b, w_hh_b, b_b, hout);
  attn_logits_kernel<<<(B_ * T_) / 256, 256, 0, stream>>>(hout, attn_w, attn_b,
                                                          logits);
  pool_ln_fc_kernel<<<B_, 256, 0, stream>>>(hout, logits, ln_g, ln_b,
                                            fc_w, fc_b, out);
}
